// RBFConv2d_67053029425167
// MI455X (gfx1250) — matrix operand order repaired and checked by reading, not compiled, not run on hardware
//
#include <hip/hip_runtime.h>

typedef __attribute__((ext_vector_type(16))) _Float16 v16h;
typedef __attribute__((ext_vector_type(8)))  _Float16 v8h;
typedef __attribute__((ext_vector_type(8)))  float    v8f;

#define IH 128
#define IW 128
#define OH 122
#define OW 122
#define NF 256
#define KK 49
#define KP 64          // K padded to 2 x 32 for wmma 16x16x32
#define XT_W 24        // padded LDS row (7 x 22 tile, pad to 24)

// ---------------------------------------------------------------------------
// Kernel 1: build augmented weight matrix W' [256][64] f16 in scratch.
//   W'[o][k] = -2*w[o][k]       k < 49
//   W'[o][49] = sum_k w[o][k]^2
//   W'[o][50..63] = 0
// ---------------------------------------------------------------------------
__global__ __launch_bounds__(256) void rbf_prep_weights(
    const float* __restrict__ w, _Float16* __restrict__ wp)
{
    int o = blockIdx.x * blockDim.x + threadIdx.x;
    if (o >= NF) return;
    const float* wr = w + (size_t)o * KK;
    _Float16 tmp[KP];
    float w2 = 0.f;
#pragma unroll
    for (int k = 0; k < KK; ++k) {
        float v = wr[k];
        w2 += v * v;
        tmp[k] = (_Float16)(-2.0f * v);
    }
    tmp[KK] = (_Float16)w2;
#pragma unroll
    for (int k = KK + 1; k < KP; ++k) tmp[k] = (_Float16)0.0f;
    _Float16* dst = wp + (size_t)o * KP;
#pragma unroll
    for (int k = 0; k < KP; ++k) dst[k] = tmp[k];
}

// ---------------------------------------------------------------------------
// Kernel 2: implicit-GEMM RBF conv via v_wmma_f32_16x16x32_f16.
// Block = 256 threads (8 waves). Block handles (n, oh, ow0..ow0+15) for all
// 256 filters: wave w computes filter tiles [32w..32w+15] and [32w+16..32w+31].
// ---------------------------------------------------------------------------
__global__ __launch_bounds__(256) void rbf_conv_wmma(
    const float* __restrict__ x, const _Float16* __restrict__ wp,
    const float* __restrict__ stdp, float* __restrict__ out)
{
    __shared__ float xt[7 * XT_W];   // 7 x 22 input tile (f32)
    __shared__ float x2s[16];        // per-output-column ||x||^2

    const int ow0  = blockIdx.x * 16;
    const int oh   = blockIdx.y;
    const int n    = blockIdx.z;

    const int tid  = threadIdx.x;
    const int lane = tid & 31;
    const int wave = tid >> 5;
    const int nn   = lane & 15;      // N index within tile
    const int hi   = lane >> 4;      // half-wave select

    // ---- cooperative load of the 7x22 x-region into LDS ----
    if (tid < 7 * 22) {
        int r = tid / 22, c = tid % 22;
        int gcol = ow0 + c;
        float v = 0.0f;
        if (gcol < IW)
            v = x[((size_t)n * IH + (size_t)(oh + r)) * IW + gcol];
        xt[r * XT_W + c] = v;
    }
    __syncthreads();

    // ---- per-column ||x||^2 (f32, threads 0..15) ----
    if (tid < 16) {
        float s = 0.0f;
#pragma unroll
        for (int r = 0; r < 7; ++r)
#pragma unroll
            for (int c = 0; c < 7; ++c) {
                float v = xt[r * XT_W + tid + c];
                s += v * v;
            }
        x2s[tid] = s;
    }

    // ---- build B fragments (patch columns), CDNA5 16-bit B layout:
    //      element e of lane l holds K = 8*hi + e (e<8) or 16 + 8*hi + (e-8) (e>=8), N = l&15 (per K-block) ----
    v16h b0, b1;
#pragma unroll
    for (int e = 0; e < 16; ++e) {
        int k0 = (e < 8) ? (hi * 8 + e) : (16 + hi * 8 + (e - 8));        // 0..31  (always < 49)
        int k1 = 32 + k0;            // 32..63
        b0[e] = (_Float16)xt[(k0 / 7) * XT_W + nn + (k0 % 7)];
        _Float16 v1;
        if (k1 < KK)       v1 = (_Float16)xt[(k1 / 7) * XT_W + nn + (k1 % 7)];
        else if (k1 == KK) v1 = (_Float16)1.0f;
        else               v1 = (_Float16)0.0f;
        b1[e] = v1;
    }

    // ---- load A fragments, CDNA5 16-bit A layout:
    //      lane l, elems 0..7  -> K = klo..klo+7
    //      lane l, elems 8..15 -> K = klo+16..klo+23,  klo = 8*hi ----
    const int klo = hi * 8;
    const int m0  = wave * 32;
    v16h a00, a01, a10, a11;
    {
        const _Float16* base  = wp + (size_t)(m0 + nn) * KP;
        const _Float16* base2 = base + 16 * KP;
        v8h t;
        t = *(const v8h*)(base  + klo);           for (int i = 0; i < 8; ++i) a00[i]     = t[i];
        t = *(const v8h*)(base  + klo + 16);      for (int i = 0; i < 8; ++i) a00[i + 8] = t[i];
        t = *(const v8h*)(base  + 32 + klo);      for (int i = 0; i < 8; ++i) a01[i]     = t[i];
        t = *(const v8h*)(base  + 32 + klo + 16); for (int i = 0; i < 8; ++i) a01[i + 8] = t[i];
        t = *(const v8h*)(base2 + klo);           for (int i = 0; i < 8; ++i) a10[i]     = t[i];
        t = *(const v8h*)(base2 + klo + 16);      for (int i = 0; i < 8; ++i) a10[i + 8] = t[i];
        t = *(const v8h*)(base2 + 32 + klo);      for (int i = 0; i < 8; ++i) a11[i]     = t[i];
        t = *(const v8h*)(base2 + 32 + klo + 16); for (int i = 0; i < 8; ++i) a11[i + 8] = t[i];
    }

    // ---- WMMA: accumulate -2*x.w + ||w||^2 over K = 64 (f32 accum) ----
    v8f c0 = {};
    v8f c1 = {};
    c0 = __builtin_amdgcn_wmma_f32_16x16x32_f16(false, a00, false, b0, (short)0, c0, false, false);
    c0 = __builtin_amdgcn_wmma_f32_16x16x32_f16(false, a01, false, b1, (short)0, c0, false, false);
    c1 = __builtin_amdgcn_wmma_f32_16x16x32_f16(false, a10, false, b0, (short)0, c1, false, false);
    c1 = __builtin_amdgcn_wmma_f32_16x16x32_f16(false, a11, false, b1, (short)0, c1, false, false);

    __syncthreads();   // make x2s visible to all waves

    // ---- epilogue: sq = max(x2 - 2x.w + w2, 0); out = exp2(coef * sq) ----
    const float x2v  = x2s[nn];
    const float stdv = stdp[0];
    // exp(-sq/(2 std^2)) = exp2(-sq * log2(e) / (2 std^2))
    const float coef = -0.7213475204444817f / (stdv * stdv);

    const bool valid = (ow0 + nn) < OW;
    const size_t plane = (size_t)OH * OW;
    const size_t obase = (((size_t)n * NF) * OH + (size_t)oh) * OW + (size_t)(ow0 + nn);

    if (valid) {
        // C/D layout: VGPR r -> M = r (lanes 0-15) or r+8 (lanes 16-31), N = lane&15
#pragma unroll
        for (int r = 0; r < 8; ++r) {
            float sq = c0[r] + x2v;
            sq = fmaxf(sq, 0.0f);
            int m = m0 + r + hi * 8;
            out[obase + (size_t)m * plane] = __builtin_exp2f(sq * coef);
        }
#pragma unroll
        for (int r = 0; r < 8; ++r) {
            float sq = c1[r] + x2v;
            sq = fmaxf(sq, 0.0f);
            int m = m0 + 16 + r + hi * 8;
            out[obase + (size_t)m * plane] = __builtin_exp2f(sq * coef);
        }
    }
}

extern "C" void kernel_launch(void* const* d_in, const int* in_sizes, int n_in,
                              void* d_out, int out_size, void* d_ws, size_t ws_size,
                              hipStream_t stream) {
    const float* x   = (const float*)d_in[0];   // (16,1,128,128)
    const float* w   = (const float*)d_in[1];   // (256,1,7,7)
    const float* std = (const float*)d_in[2];   // scalar
    float* out       = (float*)d_out;           // (16,256,122,122)
    _Float16* wp     = (_Float16*)d_ws;         // 256*64 f16 = 32 KB scratch

    rbf_prep_weights<<<1, 256, 0, stream>>>(w, wp);

    dim3 grid((OW + 15) / 16, OH, 16);          // 8 x 122 x 16 blocks
    rbf_conv_wmma<<<grid, 256, 0, stream>>>(x, wp, std, out);
}
